// LongformerWithNodeEmbeddings_16106127360280
// MI455X (gfx1250) — matrix operand order repaired and checked by reading, not compiled, not run on hardware
//
#include <hip/hip_runtime.h>
#include <cmath>

// ---------------------------------------------------------------------------
// Types / WMMA helpers (gfx1250 wave32, v_wmma_f32_16x16x32_f16)
// ---------------------------------------------------------------------------
typedef __attribute__((ext_vector_type(16))) _Float16 v16h;
typedef __attribute__((ext_vector_type(8)))  float    v8f;

__device__ inline v8f wmma16(v16h a, v16h b, v8f c) {
  // D = A(16x32 f16) * B(32x16 f16) + C(16x16 f32)
  return __builtin_amdgcn_wmma_f32_16x16x32_f16(
      /*neg_a=*/false, a, /*neg_b=*/false, b,
      /*c_mod=*/(short)0, c, /*reuse_a=*/false, /*reuse_b=*/false);
}

// A-matrix (16x32, f16) per-lane fragment. Lane (half*16+m) holds row m.
// Element i<8 -> K = kbase + half*8 + i ; i>=8 -> K = kbase + 16 + half*8 + (i-8)
// => two contiguous 8x f16 (16B) slices.
__device__ inline v16h load_a_frag(const _Float16* row, int half) {
  v16h v;
  const _Float16* p0 = row + half * 8;
  const _Float16* p1 = row + 16 + half * 8;
#pragma unroll
  for (int i = 0; i < 8; ++i) { v[i] = p0[i]; v[i + 8] = p1[i]; }
  return v;
}

// B-matrix (32x16, f16) per-lane fragment. Lane (half*16+n) holds column n.
// Element i<8 -> K = kbase + half*8 + i ; i>=8 -> K = kbase + 16 + half*8 + (i-8)
// `colrow` points at the start (k=kbase) of the source row that forms column n.
__device__ inline v16h load_b_frag(const _Float16* colrow, int half) {
  v16h v;
  const _Float16* p0 = colrow + half * 8;
  const _Float16* p1 = colrow + 16 + half * 8;
#pragma unroll
  for (int i = 0; i < 8; ++i) { v[i] = p0[i]; v[i + 8] = p1[i]; }
  return v;
}

__device__ inline float gelu_f(float x) {
  return 0.5f * x * (1.0f + erff(x * 0.70710678118654752f));
}

#define NEG_INF (-__builtin_inff())

// Problem constants
#define BATCH   32
#define HID     768
#define SEQLEN  4096
#define NHEADS  4
#define HD      64
#define N1ROWS  256       // rows of n1
#define N2REAL  257       // real rows of n2
#define N2PAD   288       // padded to 9*32 for K-chunking
#define CIN     128
#define COUT    256
#define RSPLIT  8         // row-tile splits per (b,h) in attention

// ---------------------------------------------------------------------------
// Kernel 1: data staging (transposes + f16 conversion + weight pre-transpose)
// ---------------------------------------------------------------------------
#define N1T (BATCH * N1ROWS * CIN)                 // 1048576
#define N2T (BATCH * N2PAD * CIN)                  // 1179648
#define WT  (COUT * CIN)                           // 32768
#define PREP_TOTAL (N1T + N2T + 3 * WT)            // 2326528

__global__ __launch_bounds__(256) void prep_kernel(
    const float* __restrict__ seq, const float* __restrict__ node1,
    const float* __restrict__ node2, const float* __restrict__ Wq,
    const float* __restrict__ Wk, const float* __restrict__ Wv,
    _Float16* __restrict__ n1h, _Float16* __restrict__ n2h,
    _Float16* __restrict__ wqt, _Float16* __restrict__ wkt,
    _Float16* __restrict__ wvt) {
  size_t i = (size_t)blockIdx.x * 256 + threadIdx.x;
  if (i >= PREP_TOTAL) return;
  if (i < N1T) {
    // n1[b][j][c] = node1[b][c][j]
    size_t b = i / (N1ROWS * CIN), rem = i % (N1ROWS * CIN);
    int j = (int)(rem / CIN), c = (int)(rem % CIN);
    n1h[i] = (_Float16)node1[b * (size_t)(CIN * N1ROWS) + (size_t)c * N1ROWS + j];
  } else if (i < (size_t)N1T + N2T) {
    size_t i2 = i - N1T;
    size_t b = i2 / (N2PAD * CIN), rem = i2 % (N2PAD * CIN);
    int j = (int)(rem / CIN), c = (int)(rem % CIN);
    float v;
    if (j < 256)       v = node2[b * (size_t)(CIN * N1ROWS) + (size_t)c * N1ROWS + j];
    else if (j == 256) v = seq[b * (size_t)SEQLEN * HID + c];  // pooled[:, :128]
    else               v = 0.0f;                               // zero pad
    n2h[i2] = (_Float16)v;
  } else {
    size_t i3 = i - N1T - N2T;
    int w = (int)(i3 / WT);
    int rem = (int)(i3 % WT);
    int o = rem / CIN, c = rem % CIN;  // transposed: Wt[o][c] = W[c][o]
    const float* Wsrc = (w == 0) ? Wq : (w == 1) ? Wk : Wv;
    _Float16* dst = (w == 0) ? wqt : (w == 1) ? wkt : wvt;
    dst[rem] = (_Float16)Wsrc[(size_t)c * COUT + o];
  }
}

// ---------------------------------------------------------------------------
// Kernel 2: projections. One wave = one 16x16 output tile, K=128 (4 WMMAs).
// Outputs: q1h (B,256,256) f16, k2h (B,288,256) f16,
//          gv1f = gelu(v1) (B,256,256) f32, gv2t = gelu(v2)^T (B,256,288) f16
// ---------------------------------------------------------------------------
#define TILES_PER_B (2 * (N1ROWS / 16) * (COUT / 16) + 2 * (N2PAD / 16) * (COUT / 16)) // 1088
#define PROJ_WAVES  (BATCH * TILES_PER_B)   // 34816  (= 4352 blocks * 8 waves)

__global__ __launch_bounds__(256) void proj_kernel(
    const _Float16* __restrict__ n1h, const _Float16* __restrict__ n2h,
    const _Float16* __restrict__ wqt, const _Float16* __restrict__ wkt,
    const _Float16* __restrict__ wvt, const float* __restrict__ bq,
    const float* __restrict__ bk, const float* __restrict__ bv,
    _Float16* __restrict__ q1h, _Float16* __restrict__ k2h,
    float* __restrict__ gv1f, _Float16* __restrict__ gv2t) {
  int wave = blockIdx.x * 8 + (threadIdx.x >> 5);
  if (wave >= PROJ_WAVES) return;
  int lane = threadIdx.x & 31;
  int ln   = lane & 15;
  int half = lane >> 4;

  int b = wave / TILES_PER_B;
  int r = wave % TILES_PER_B;

  const _Float16* A;
  const _Float16* W;
  const float* bs;
  _Float16* dst16 = nullptr;
  float*    dstf  = nullptr;
  int rt, ct, mode;  // mode 0: f16 row-major(256); 1: f32 gelu row-major; 2: f16 gelu transposed(288)
  if (r < 512) {
    int which = r >> 8;
    int tile = r & 255;
    rt = tile >> 4; ct = tile & 15;
    A = n1h + (size_t)b * N1ROWS * CIN;
    if (which == 0) { W = wqt; bs = bq; mode = 0; dst16 = q1h + (size_t)b * N1ROWS * COUT; }
    else            { W = wvt; bs = bv; mode = 1; dstf  = gv1f + (size_t)b * N1ROWS * COUT; }
  } else {
    int r2 = r - 512;
    int which = r2 / 288;
    int tile = r2 % 288;
    rt = tile / 16; ct = tile % 16;
    A = n2h + (size_t)b * N2PAD * CIN;
    if (which == 0) { W = wkt; bs = bk; mode = 0; dst16 = k2h + (size_t)b * N2PAD * COUT; }
    else            { W = wvt; bs = bv; mode = 2; dst16 = gv2t + (size_t)b * COUT * N2PAD; }
  }

  const _Float16* arow = A + (size_t)(rt * 16 + ln) * CIN;
  const _Float16* brow = W + (size_t)(ct * 16 + ln) * CIN;

  v8f acc = {};
#pragma unroll
  for (int kc = 0; kc < 4; ++kc) {
    int kb = kc * 32;
    acc = wmma16(load_a_frag(arow + kb, half), load_b_frag(brow + kb, half), acc);
  }

  float bval = bs[ct * 16 + ln];  // C layout: lane holds column n = ct*16+ln
#pragma unroll
  for (int r8 = 0; r8 < 8; ++r8) {
    int m = rt * 16 + r8 + 8 * half;
    int o = ct * 16 + ln;
    float v = acc[r8] + bval;
    if (mode == 0)      dst16[(size_t)m * COUT + o] = (_Float16)v;
    else if (mode == 1) dstf [(size_t)m * COUT + o] = gelu_f(v);
    else                dst16[(size_t)o * N2PAD + m] = (_Float16)gelu_f(v);
  }
}

// ---------------------------------------------------------------------------
// Kernel 3: fused attention + gelu(v1) add + partial max-pool over rows.
// Grid = B*H*RSPLIT blocks; each block (2 waves) owns 2 row-tiles of 16 q-rows.
// Writes per-(b,h,rsplit) partial column maxima; kernel 3b reduces them.
// ---------------------------------------------------------------------------
#define SROWLEN 292  // 288 + pad

__global__ __launch_bounds__(64) void attn_kernel(
    const _Float16* __restrict__ q1h, const _Float16* __restrict__ k2h,
    const _Float16* __restrict__ gv2t, const float* __restrict__ gv1f,
    float* __restrict__ partial) {
  __shared__ float    S[2][16][SROWLEN];
  __shared__ _Float16 P[2][16][SROWLEN];
  __shared__ float    wmax[2][64];

  int w    = threadIdx.x >> 5;
  int lane = threadIdx.x & 31;
  int ln   = lane & 15;
  int half = lane >> 4;
  int bh     = blockIdx.x / RSPLIT;   // (b*4 + h)
  int rsplit = blockIdx.x % RSPLIT;
  int b = bh >> 2;
  int h = bh & 3;
  int rt = rsplit * 2 + w;            // this wave's row tile: q rows [rt*16, rt*16+16)

  float rm[4] = {NEG_INF, NEG_INF, NEG_INF, NEG_INF};

  // Q fragments for this row-tile (K = 64 head dims -> 2 chunks)
  const _Float16* qrow = q1h + ((size_t)b * N1ROWS + rt * 16 + ln) * COUT + h * HD;
  v16h aq0 = load_a_frag(qrow, half);
  v16h aq1 = load_a_frag(qrow + 32, half);

  // scores = Q K2^T / 8 into LDS, masked beyond the 257 real keys
  for (int t = 0; t < N2PAD / 16; ++t) {
    const _Float16* krow = k2h + ((size_t)b * N2PAD + t * 16 + ln) * COUT + h * HD;
    v8f c = {};
    c = wmma16(aq0, load_b_frag(krow, half), c);
    c = wmma16(aq1, load_b_frag(krow + 32, half), c);
    int j = t * 16 + ln;
#pragma unroll
    for (int r8 = 0; r8 < 8; ++r8)
      S[w][r8 + 8 * half][j] = (j < N2REAL) ? c[r8] * 0.125f : NEG_INF;
  }
  __syncthreads();

  // row softmax (16 rows handled by lanes 0..15 of each wave)
  if (lane < 16) {
    float mx = NEG_INF;
    for (int j = 0; j < N2PAD; ++j) mx = fmaxf(mx, S[w][lane][j]);
    float sum = 0.0f;
    for (int j = 0; j < N2PAD; ++j) {
      float e = expf(S[w][lane][j] - mx);
      S[w][lane][j] = e;
      sum += e;
    }
    float inv = 1.0f / sum;
    for (int j = 0; j < N2PAD; ++j)
      P[w][lane][j] = (_Float16)(S[w][lane][j] * inv);
  }
  __syncthreads();

  // ctx = P (16x288) @ gelu(v2) (288x64); + gelu(v1); column max over 16 rows
  const _Float16* prow = &P[w][ln][0];
#pragma unroll
  for (int nt = 0; nt < 4; ++nt) {
    const _Float16* grow = gv2t + ((size_t)b * COUT + h * HD + nt * 16 + ln) * N2PAD;
    v8f acc = {};
#pragma unroll
    for (int kc = 0; kc < N2PAD / 32; ++kc) {
      int kb = kc * 32;
      acc = wmma16(load_a_frag(prow + kb, half), load_b_frag(grow + kb, half), acc);
    }
    float lm = NEG_INF;
#pragma unroll
    for (int r8 = 0; r8 < 8; ++r8) {
      int m = rt * 16 + r8 + 8 * half;
      int o = h * HD + nt * 16 + ln;
      float v = acc[r8] + gv1f[((size_t)b * N1ROWS + m) * COUT + o];
      lm = fmaxf(lm, v);
    }
    rm[nt] = fmaxf(rm[nt], lm);
  }

  // combine the two half-row groups (lanes n and n+16 hold the same column)
#pragma unroll
  for (int nt = 0; nt < 4; ++nt)
    rm[nt] = fmaxf(rm[nt], __shfl_xor(rm[nt], 16, 32));
  if (lane < 16) {
#pragma unroll
    for (int nt = 0; nt < 4; ++nt) wmax[w][nt * 16 + ln] = rm[nt];
  }
  __syncthreads();
  if (threadIdx.x < 64) {
    float v = fmaxf(wmax[0][threadIdx.x], wmax[1][threadIdx.x]);
    partial[((size_t)bh * RSPLIT + rsplit) * 64 + threadIdx.x] = v;
  }
}

// ---------------------------------------------------------------------------
// Kernel 3b: reduce partial maxima -> concat vector (B,256)
// ---------------------------------------------------------------------------
__global__ __launch_bounds__(256) void reduce_max_kernel(
    const float* __restrict__ partial, float* __restrict__ concatv) {
  int u = blockIdx.x * 256 + threadIdx.x;  // 0 .. B*256-1 ; u = b*256 + h*64 + o
  if (u >= BATCH * 256) return;
  int b = u >> 8;
  int c = u & 255;
  int h = c >> 6;
  int o = c & 63;
  float m = NEG_INF;
#pragma unroll
  for (int r = 0; r < RSPLIT; ++r)
    m = fmaxf(m, partial[(((size_t)(b * 4 + h) * RSPLIT) + r) * 64 + o]);
  concatv[u] = m;
}

// ---------------------------------------------------------------------------
// Kernel 4: head (final_ctx, layernorm, fc1/fc2/fc3). One block per batch row.
// ---------------------------------------------------------------------------
__global__ __launch_bounds__(256) void head_kernel(
    const float* __restrict__ seq, const float* __restrict__ concatv,
    const float* __restrict__ Wo, const float* __restrict__ bo,
    const float* __restrict__ ln_g, const float* __restrict__ ln_b,
    const float* __restrict__ fc1_w, const float* __restrict__ fc1_b,
    const float* __restrict__ fc2_w, const float* __restrict__ fc2_b,
    const float* __restrict__ fc3_w, const float* __restrict__ fc3_b,
    float* __restrict__ out) {
  __shared__ float emb[HID + 64];
  __shared__ float h1[512];
  __shared__ float h2[200];
  __shared__ float red[256];
  int t = threadIdx.x;
  int b = blockIdx.x;
  const int E = HID + 64;  // 832

  // final_ctx = concat @ Wo + bo  -> emb[768..831]
  if (t < 64) {
    float s = bo[t];
    for (int c = 0; c < 256; ++c) s += concatv[b * 256 + c] * Wo[(size_t)c * 64 + t];
    emb[HID + t] = s;
  }
  for (int i = t; i < HID; i += 256) emb[i] = seq[(size_t)b * SEQLEN * HID + i];
  __syncthreads();

  // mean
  float ps = 0.0f;
  for (int i = t; i < E; i += 256) ps += emb[i];
  red[t] = ps;
  __syncthreads();
  if (t == 0) { float tot = 0.0f; for (int k = 0; k < 256; ++k) tot += red[k]; red[0] = tot; }
  __syncthreads();
  float mu = red[0] / (float)E;
  __syncthreads();
  // variance (population)
  float ps2 = 0.0f;
  for (int i = t; i < E; i += 256) { float d = emb[i] - mu; ps2 += d * d; }
  red[t] = ps2;
  __syncthreads();
  if (t == 0) { float tot = 0.0f; for (int k = 0; k < 256; ++k) tot += red[k]; red[0] = tot; }
  __syncthreads();
  float inv_sd = rsqrtf(red[0] / (float)E + 1e-5f);
  __syncthreads();
  for (int i = t; i < E; i += 256)
    emb[i] = (emb[i] - mu) * inv_sd * ln_g[i] + ln_b[i];
  __syncthreads();

  // fc1: 832 -> 512, gelu
  for (int j = t; j < 512; j += 256) {
    float s = fc1_b[j];
    for (int c = 0; c < E; ++c) s += emb[c] * fc1_w[(size_t)c * 512 + j];
    h1[j] = gelu_f(s);
  }
  __syncthreads();
  // fc2: 512 -> 200, gelu
  for (int j = t; j < 200; j += 256) {
    float s = fc2_b[j];
    for (int c = 0; c < 512; ++c) s += h1[c] * fc2_w[(size_t)c * 200 + j];
    h2[j] = gelu_f(s);
  }
  __syncthreads();
  // fc3: 200 -> 2
  if (t < 2) {
    float s = fc3_b[t];
    for (int c = 0; c < 200; ++c) s += h2[c] * fc3_w[(size_t)c * 2 + t];
    out[b * 2 + t] = s;
  }
}

// ---------------------------------------------------------------------------
// Launch
// ---------------------------------------------------------------------------
extern "C" void kernel_launch(void* const* d_in, const int* in_sizes, int n_in,
                              void* d_out, int out_size, void* d_ws, size_t ws_size,
                              hipStream_t stream) {
  const float* seq   = (const float*)d_in[0];
  const float* node1 = (const float*)d_in[1];
  const float* node2 = (const float*)d_in[2];
  const float* Wq    = (const float*)d_in[3];
  const float* bq    = (const float*)d_in[4];
  const float* Wk    = (const float*)d_in[5];
  const float* bk    = (const float*)d_in[6];
  const float* Wv    = (const float*)d_in[7];
  const float* bv    = (const float*)d_in[8];
  const float* Wo    = (const float*)d_in[9];
  const float* bo    = (const float*)d_in[10];
  const float* ln_g  = (const float*)d_in[11];
  const float* ln_b  = (const float*)d_in[12];
  const float* fc1_w = (const float*)d_in[13];
  const float* fc1_b = (const float*)d_in[14];
  const float* fc2_w = (const float*)d_in[15];
  const float* fc2_b = (const float*)d_in[16];
  const float* fc3_w = (const float*)d_in[17];
  const float* fc3_b = (const float*)d_in[18];
  float* out = (float*)d_out;

  // Workspace carving (256B aligned)
  char* p = (char*)d_ws;
  auto take = [&](size_t bytes) {
    void* r = (void*)p;
    p += (bytes + 255) & ~(size_t)255;
    return r;
  };
  _Float16* n1h   = (_Float16*)take((size_t)N1T * 2);
  _Float16* n2h   = (_Float16*)take((size_t)N2T * 2);
  _Float16* wqt   = (_Float16*)take((size_t)WT * 2);
  _Float16* wkt   = (_Float16*)take((size_t)WT * 2);
  _Float16* wvt   = (_Float16*)take((size_t)WT * 2);
  _Float16* q1h   = (_Float16*)take((size_t)BATCH * N1ROWS * COUT * 2);
  _Float16* k2h   = (_Float16*)take((size_t)BATCH * N2PAD * COUT * 2);
  float*    gv1f  = (float*)   take((size_t)BATCH * N1ROWS * COUT * 4);
  _Float16* gv2t  = (_Float16*)take((size_t)BATCH * COUT * N2PAD * 2);
  float*    part  = (float*)   take((size_t)BATCH * NHEADS * RSPLIT * 64 * 4);
  float*    cconc = (float*)   take((size_t)BATCH * 256 * 4);

  prep_kernel<<<(PREP_TOTAL + 255) / 256, 256, 0, stream>>>(
      seq, node1, node2, Wq, Wk, Wv, n1h, n2h, wqt, wkt, wvt);

  proj_kernel<<<PROJ_WAVES / 8, 256, 0, stream>>>(
      n1h, n2h, wqt, wkt, wvt, bq, bk, bv, q1h, k2h, gv1f, gv2t);

  attn_kernel<<<BATCH * NHEADS * RSPLIT, 64, 0, stream>>>(
      q1h, k2h, gv2t, gv1f, part);

  reduce_max_kernel<<<(BATCH * 256 + 255) / 256, 256, 0, stream>>>(part, cconc);

  head_kernel<<<BATCH, 256, 0, stream>>>(
      seq, cconc, Wo, bo, ln_g, ln_b, fc1_w, fc1_b, fc2_w, fc2_b, fc3_w, fc3_b, out);
}
